// VQVAEEncoder_4260607558144
// MI455X (gfx1250) — matrix operand order repaired and checked by reading, not compiled, not run on hardware
//
#include <hip/hip_runtime.h>
#include <hip/hip_bf16.h>
#include <math.h>

typedef __attribute__((ext_vector_type(16))) _Float16 v16h;
typedef __attribute__((ext_vector_type(8)))  _Float16 v8h;
typedef __attribute__((ext_vector_type(8)))  float    v8f;

#define NEGBIG (-1000000000.0f)

// ---------------------------------------------------------------------------
// WMMA fragment helpers (layouts per CDNA5 ISA 7.12.2, wave32)
// A 16x32 f16: lanes 0-15 -> row=lane, K = {0..7, 16..23}; lanes 16-31 -> K = {8..15, 24..31}
// B 32x16 f16 (stored transposed in LDS: Bt[col][k]): lanes 0-15 -> col=lane, K = {0..7, 16..23};
//                                                      lanes 16-31 -> col=lane-16, K = {8..15, 24..31}
// C/D 16x16 f32: VGPR r, lanes 0-15 -> (M=r, N=lane); lanes 16-31 -> (M=r+8, N=lane-16)
// ---------------------------------------------------------------------------
__device__ __forceinline__ v16h frag_a(const _Float16* p, int ld) {
  int lane = threadIdx.x & 31;
  int m = lane & 15;
  int hh = lane >> 4;
  const _Float16* r = p + m * ld + hh * 8;
  v8h lo = *(const v8h*)(r);
  v8h hi = *(const v8h*)(r + 16);
  v16h o;
#pragma unroll
  for (int i = 0; i < 8; ++i) { o[i] = lo[i]; o[i + 8] = hi[i]; }
  return o;
}

__device__ __forceinline__ v16h frag_b(const _Float16* p, int ld) {
  int lane = threadIdx.x & 31;
  int n = lane & 15;
  int hh = lane >> 4;
  const _Float16* r = p + n * ld + hh * 8;
  v8h lo = *(const v8h*)(r);
  v8h hi = *(const v8h*)(r + 16);
  v16h o;
#pragma unroll
  for (int i = 0; i < 8; ++i) { o[i] = lo[i]; o[i + 8] = hi[i]; }
  return o;
}

__device__ __forceinline__ v8f wmma16(v16h a, v16h b, v8f c) {
  return __builtin_amdgcn_wmma_f32_16x16x32_f16(false, a, false, b, (short)0, c, false, false);
}

// ---------------------------------------------------------------------------
// Conversions
// ---------------------------------------------------------------------------
__global__ void cvt_f32_f16(const float* __restrict__ src, _Float16* __restrict__ dst, int n) {
  int i = blockIdx.x * 256 + threadIdx.x;
  if (i < n) dst[i] = (_Float16)src[i];
}

// Transpose one or more [K,N] f32 weight slices into [N,K] f16 (blockIdx.y = slice).
__global__ void transpose_f32_f16(const float* __restrict__ src, _Float16* __restrict__ dst,
                                  int K, int N) {
  size_t lay = blockIdx.y;
  const float* s = src + lay * (size_t)K * N;
  _Float16*    d = dst + lay * (size_t)K * N;
  int i = blockIdx.x * 256 + threadIdx.x;
  if (i < K * N) {
    int k = i / N, n = i % N;
    d[(size_t)n * K + k] = (_Float16)s[i];  // coalesced read, scattered one-time write
  }
}

__global__ void zero_f32(float* __restrict__ p, int n) {
  int i = blockIdx.x * 256 + threadIdx.x;
  if (i < n) p[i] = 0.0f;
}

__global__ void nvalid_kernel(const int* __restrict__ amask, float* __restrict__ outv) {
  __shared__ float red[256];
  int tid = threadIdx.x;
  float s = 0.0f;
  for (int i = tid; i < 8192; i += 256) s += (float)amask[i];
  red[tid] = s; __syncthreads();
  for (int off = 128; off > 0; off >>= 1) { if (tid < off) red[tid] += red[tid + off]; __syncthreads(); }
  if (tid == 0) outv[0] = red[0];
}

// ---------------------------------------------------------------------------
// WMMA GEMM: out = act(A[M,K] @ B + bias), with B given PRE-TRANSPOSED as Bt[N,K] f16.
// Block tile 64(M) x 256(N), BK=32, 256 threads = 8 waves; wave tile 32x64.
// Staging: all 5 global v8h loads issued BEFORE the barrier into distinct temps
// (one loadcnt wait, latency overlapped with previous iteration's WMMA tail).
// ---------------------------------------------------------------------------
__global__ __launch_bounds__(256) void gemm_f16_kernel(
    const _Float16* __restrict__ A, const _Float16* __restrict__ Bt,
    const float* __restrict__ bias, float* __restrict__ out32,
    _Float16* __restrict__ out16, _Float16* __restrict__ out16t,
    int M, int N, int K, int act)
{
  __shared__ __align__(16) _Float16 Asm[64 * 40];
  __shared__ __align__(16) _Float16 Bsm[256 * 40];
  const int tid = threadIdx.x;
  const int wid = tid >> 5;
  const int rowBase = blockIdx.y * 64;
  const int colBase = blockIdx.x * 256;
  const int wm = (wid & 1) * 32;
  const int wn = (wid >> 1) * 64;
  v8f acc[2][4] = {};

  const int ar = tid >> 2, ac = (tid & 3) * 8;   // A staging: 64 rows x 32 k

  for (int k0 = 0; k0 < K; k0 += 32) {
    // ---- issue all global loads first (no LDS dependency) ----
    v8h ta = *(const v8h*)(A + (size_t)(rowBase + ar) * K + k0 + ac);
    v8h tb[4];
    int bcol[4], bc[4];
#pragma unroll
    for (int rep = 0; rep < 4; ++rep) {
      int e = rep * 2048 + tid * 8;
      bcol[rep] = e >> 5; bc[rep] = e & 31;
      tb[rep] = *(const v8h*)(Bt + (size_t)(colBase + bcol[rep]) * K + k0 + bc[rep]);
    }
    if (k0 + 32 < K) { // prefetch next K-step tiles (global_prefetch_b8)
      __builtin_prefetch(A + (size_t)(rowBase + ar) * K + k0 + 32 + ac, 0, 1);
      __builtin_prefetch(Bt + (size_t)(colBase + (tid >> 2)) * K + k0 + 32 + ((tid & 3) * 8), 0, 1);
    }
    __syncthreads();           // previous iteration's frag reads complete
    *(v8h*)(Asm + ar * 40 + ac) = ta;
#pragma unroll
    for (int rep = 0; rep < 4; ++rep)
      *(v8h*)(Bsm + bcol[rep] * 40 + bc[rep]) = tb[rep];
    __syncthreads();

    v16h a0 = frag_a(Asm + (wm + 0) * 40, 40);
    v16h a1 = frag_a(Asm + (wm + 16) * 40, 40);
    v16h bf[4];
#pragma unroll
    for (int j = 0; j < 4; ++j) bf[j] = frag_b(Bsm + (wn + j * 16) * 40, 40);
#pragma unroll
    for (int j = 0; j < 4; ++j) {
      acc[0][j] = wmma16(a0, bf[j], acc[0][j]);
      acc[1][j] = wmma16(a1, bf[j], acc[1][j]);
    }
  }

  const int lane = tid & 31;
  const int ln = lane & 15, lh = lane >> 4;
#pragma unroll
  for (int i = 0; i < 2; ++i)
#pragma unroll
    for (int j = 0; j < 4; ++j) {
      int col = colBase + wn + j * 16 + ln;
      float bvv = bias ? bias[col] : 0.0f;
      int row0 = rowBase + wm + i * 16 + lh * 8;
      float vals[8];
#pragma unroll
      for (int r = 0; r < 8; ++r) {
        float v = acc[i][j][r] + bvv;
        if (act == 1) v = 0.5f * v * (1.0f + erff(v * 0.70710678118f)); // exact GELU
        vals[r] = v;
      }
      if (out32) {
#pragma unroll
        for (int r = 0; r < 8; ++r) out32[(size_t)(row0 + r) * N + col] = vals[r];
      }
      if (out16) {
#pragma unroll
        for (int r = 0; r < 8; ++r) out16[(size_t)(row0 + r) * N + col] = (_Float16)vals[r];
      }
      if (out16t) { // transposed f16 out: 8 consecutive rows -> one v8h store
        v8h t;
#pragma unroll
        for (int r = 0; r < 8; ++r) t[r] = (_Float16)vals[r];
        *(v8h*)(out16t + (size_t)col * M + row0) = t;
      }
    }
}

// ---------------------------------------------------------------------------
// Flash attention: per block one (b, h) and 64 query rows; key chunks of 64.
// Online softmax; PV accumulator held in WMMA register fragments (3 tiles/wave).
// q,k: f16 [B,S,H*dk]; v given transposed as vT[H*dk][B*S]; ctx: f16 [B,S,H*dk].
// ---------------------------------------------------------------------------
__global__ __launch_bounds__(256) void flash_attn_kernel(
    const _Float16* __restrict__ q, const _Float16* __restrict__ k,
    const _Float16* __restrict__ vT, const int* __restrict__ amask,
    _Float16* __restrict__ ctx)
{
  __shared__ __align__(16) _Float16 Qs[64 * 96];
  __shared__ __align__(16) _Float16 Ks[64 * 96];
  __shared__ __align__(16) _Float16 Vt[96 * 64];   // Vt[d][key]
  __shared__ __align__(16) float    Ssm[64 * 64];
  __shared__ __align__(16) _Float16 Ps[64 * 64];
  __shared__ float alpha_s[64];
  __shared__ float rowl_s[64];
  __shared__ int   msk[64];

  const int tid = threadIdx.x;
  const int wid = tid >> 5;
  const int lane = tid & 31;
  const int ln = lane & 15, lh = lane >> 4;
  const int b = blockIdx.z, h = blockIdx.y;
  const int q0 = blockIdx.x * 64;
  const size_t hoff = (size_t)h * 96;
  const float scale = 0.1020620726159658f; // 1/sqrt(96)

#pragma unroll
  for (int i = 0; i < 3; ++i) { // load Q block 64x96
    int j = tid + i * 256;
    int r = j / 12, d8 = (j % 12) * 8;
    v8h t = *(const v8h*)(q + (size_t)(b * 1024 + q0 + r) * 768 + hoff + d8);
    *(v8h*)(Qs + r * 96 + d8) = t;
  }

  float mrun = -INFINITY, lrun = 0.0f;
  v8f oacc[3] = {};
  int tmv[3], tnv[3];
#pragma unroll
  for (int t = 0; t < 3; ++t) { int lin = wid * 3 + t; tmv[t] = lin / 6; tnv[t] = lin % 6; }

  for (int kc = 0; kc < 16; ++kc) {
    int kk0 = kc * 64;
    // ---- issue all K/V global loads before the barrier ----
    v8h tk[3], tv[3];
    int kr[3], kd8[3], vdd[3], vr8[3];
#pragma unroll
    for (int i = 0; i < 3; ++i) {
      int j = tid + i * 256;
      kr[i] = j / 12; kd8[i] = (j % 12) * 8;
      tk[i] = *(const v8h*)(k + (size_t)(b * 1024 + kk0 + kr[i]) * 768 + hoff + kd8[i]);
      vdd[i] = j >> 3; vr8[i] = (j & 7) * 8;
      tv[i] = *(const v8h*)(vT + (hoff + vdd[i]) * 8192 + b * 1024 + kk0 + vr8[i]);
    }
    int mv = (tid < 64) ? amask[b * 1024 + kk0 + tid] : 0;
    __syncthreads();
#pragma unroll
    for (int i = 0; i < 3; ++i) {
      *(v8h*)(Ks + kr[i] * 96 + kd8[i]) = tk[i];
      *(v8h*)(Vt + vdd[i] * 64 + vr8[i]) = tv[i];
    }
    if (tid < 64) msk[tid] = mv;
    __syncthreads();

    { // scores 64x64: wave -> rows (wid&3)*16, cols (wid>>2)*32
      int m0 = (wid & 3) * 16;
      int n0 = (wid >> 2) * 32;
      v8f s0 = {}, s1 = {};
#pragma unroll
      for (int kk = 0; kk < 96; kk += 32) {
        v16h a  = frag_a(Qs + m0 * 96 + kk, 96);
        v16h b0 = frag_b(Ks + n0 * 96 + kk, 96);
        v16h b1 = frag_b(Ks + (n0 + 16) * 96 + kk, 96);
        s0 = wmma16(a, b0, s0);
        s1 = wmma16(a, b1, s1);
      }
#pragma unroll
      for (int r = 0; r < 8; ++r) {
        int row = m0 + lh * 8 + r;
        int c0 = n0 + ln, c1 = n0 + 16 + ln;
        float v0 = s0[r] * scale; if (msk[c0] == 0) v0 = NEGBIG;
        float v1 = s1[r] * scale; if (msk[c1] == 0) v1 = NEGBIG;
        Ssm[row * 64 + c0] = v0;
        Ssm[row * 64 + c1] = v1;
      }
    }
    __syncthreads();

    if (tid < 64) { // online softmax per row (thread = row)
      float mc = -INFINITY;
      for (int c = 0; c < 64; ++c) mc = fmaxf(mc, Ssm[tid * 64 + c]);
      float mnew = fmaxf(mrun, mc);
      float al = __expf(mrun - mnew);
      float s = 0.0f;
      for (int c = 0; c < 64; ++c) {
        float p = __expf(Ssm[tid * 64 + c] - mnew);
        Ps[tid * 64 + c] = (_Float16)p;
        s += p;
      }
      lrun = lrun * al + s;
      mrun = mnew;
      alpha_s[tid] = al;
    }
    __syncthreads();

#pragma unroll
    for (int t = 0; t < 3; ++t) { // PV: rescale accumulator then accumulate chunk
#pragma unroll
      for (int r = 0; r < 8; ++r) {
        int row = tmv[t] * 16 + lh * 8 + r;
        oacc[t][r] *= alpha_s[row];
      }
#pragma unroll
      for (int kk = 0; kk < 64; kk += 32) {
        v16h a  = frag_a(Ps + tmv[t] * 16 * 64 + kk, 64);
        v16h bb = frag_b(Vt + tnv[t] * 16 * 64 + kk, 64);
        oacc[t] = wmma16(a, bb, oacc[t]);
      }
    }
  }
  __syncthreads();
  if (tid < 64) rowl_s[tid] = lrun;
  __syncthreads();
#pragma unroll
  for (int t = 0; t < 3; ++t) {
    int d = tnv[t] * 16 + ln;
#pragma unroll
    for (int r = 0; r < 8; ++r) {
      int row = tmv[t] * 16 + lh * 8 + r;
      float val = oacc[t][r] / rowl_s[row];
      ctx[(size_t)(b * 1024 + q0 + row) * 768 + hoff + d] = (_Float16)val;
    }
  }
}

// ---------------------------------------------------------------------------
// Fused residual + layernorm: out = LN(x + delta) * g + b. One block per row, 768 cols.
// ---------------------------------------------------------------------------
__global__ __launch_bounds__(256) void ln_kernel(
    const float* __restrict__ x, const float* __restrict__ delta,
    const float* __restrict__ g, const float* __restrict__ bsh,
    float* __restrict__ out32, _Float16* __restrict__ out16)
{
  __shared__ float red[256];
  const int row = blockIdx.x;
  const int tid = threadIdx.x;
  const size_t base = (size_t)row * 768;
  float vals[3];
  float s = 0.0f;
#pragma unroll
  for (int i = 0; i < 3; ++i) {
    int c = tid + i * 256;
    float vv = x[base + c];
    if (delta) vv += delta[base + c];
    vals[i] = vv;
    s += vv;
  }
  red[tid] = s; __syncthreads();
  for (int off = 128; off > 0; off >>= 1) { if (tid < off) red[tid] += red[tid + off]; __syncthreads(); }
  float mean = red[0] * (1.0f / 768.0f);
  __syncthreads();
  float sq = 0.0f;
#pragma unroll
  for (int i = 0; i < 3; ++i) { float d = vals[i] - mean; sq += d * d; }
  red[tid] = sq; __syncthreads();
  for (int off = 128; off > 0; off >>= 1) { if (tid < off) red[tid] += red[tid + off]; __syncthreads(); }
  float inv = rsqrtf(red[0] * (1.0f / 768.0f) + 1e-5f);
#pragma unroll
  for (int i = 0; i < 3; ++i) {
    int c = tid + i * 256;
    float o = (vals[i] - mean) * inv * g[c] + bsh[c];
    out32[base + c] = o;
    if (out16) out16[base + c] = (_Float16)o;
  }
}

// ---------------------------------------------------------------------------
// VQ kernel: per block 64 query rows vs all 8192 codebook keys (two-pass recompute).
// ---------------------------------------------------------------------------
__global__ __launch_bounds__(256) void vq_kernel(
    const _Float16* __restrict__ qy, const _Float16* __restrict__ ky,
    const float* __restrict__ hq, const float* __restrict__ emb,
    const int* __restrict__ amask, const float* __restrict__ temp,
    float* __restrict__ quant_out, float* __restrict__ idx_out,
    float* __restrict__ avg_glob, float* __restrict__ mse_glob)
{
  __shared__ __align__(16) _Float16 Qs[64 * 40];
  __shared__ __align__(16) _Float16 Ks[64 * 40];
  __shared__ __align__(16) float    Ssm[64 * 64];
  __shared__ float avg[8192];
  __shared__ float rowm_s[64], rowl_s[64];
  __shared__ int   rmask_s[64];

  const int tid = threadIdx.x;
  const int wid = tid >> 5;
  const int lane = tid & 31;
  const int ln = lane & 15, lh = lane >> 4;
  const int r0 = blockIdx.x * 64;
  const float invt = 1.0f / temp[0];

  for (int i = tid; i < 8192; i += 256) avg[i] = 0.0f;
  if (tid < 64) rmask_s[tid] = amask[r0 + tid];

  float mrun = -INFINITY, lrun = 0.0f, bestv = -INFINITY;
  int besti = 0;

  const int m0 = (wid & 3) * 16;
  const int n0 = (wid >> 2) * 32;
  const int sr = tid >> 2, sc = (tid & 3) * 8;   // staging: 64 rows x 32 k

  for (int pass = 0; pass < 2; ++pass) {
    for (int cb = 0; cb < 128; ++cb) {
      int col0 = cb * 64;
      v8f s0 = {}, s1 = {};
      for (int k0 = 0; k0 < 768; k0 += 32) {
        // issue both global loads before the barrier
        v8h tq = *(const v8h*)(qy + (size_t)(r0 + sr) * 768 + k0 + sc);
        v8h tk = *(const v8h*)(ky + (size_t)(col0 + sr) * 768 + k0 + sc);
        __syncthreads();
        *(v8h*)(Qs + sr * 40 + sc) = tq;
        *(v8h*)(Ks + sr * 40 + sc) = tk;
        __syncthreads();
        v16h a  = frag_a(Qs + m0 * 40, 40);
        v16h b0 = frag_b(Ks + n0 * 40, 40);
        v16h b1 = frag_b(Ks + (n0 + 16) * 40, 40);
        s0 = wmma16(a, b0, s0);
        s1 = wmma16(a, b1, s1);
      }
      __syncthreads();
#pragma unroll
      for (int r = 0; r < 8; ++r) {
        int row = m0 + lh * 8 + r;
        Ssm[row * 64 + n0 + ln]      = s0[r] * invt;
        Ssm[row * 64 + n0 + 16 + ln] = s1[r] * invt;
      }
      __syncthreads();
      if (pass == 0) {
        if (tid < 64 && rmask_s[tid] != 0) { // thread = row: online stats + argmax
          float mc = -INFINITY;
          for (int c = 0; c < 64; ++c) {
            float s = Ssm[tid * 64 + c];
            mc = fmaxf(mc, s);
            if (s > bestv) { bestv = s; besti = col0 + c; }
          }
          float mnew = fmaxf(mrun, mc);
          float al = __expf(mrun - mnew);
          float acc = 0.0f;
          for (int c = 0; c < 64; ++c) acc += __expf(Ssm[tid * 64 + c] - mnew);
          lrun = lrun * al + acc;
          mrun = mnew;
        }
      } else {
        if (tid < 64) { // thread = column: sum softmax probs over rows
          float acc = 0.0f;
          for (int r = 0; r < 64; ++r) {
            if (rmask_s[r] != 0)
              acc += __expf(Ssm[r * 64 + tid] - rowm_s[r]) / rowl_s[r];
          }
          avg[col0 + tid] += acc;
        }
      }
      __syncthreads();
    }
    if (pass == 0) {
      if (tid < 64) { rowm_s[tid] = mrun; rowl_s[tid] = lrun; }
      __syncthreads();
    }
  }

  for (int i = tid; i < 8192; i += 256) {
    float a = avg[i];
    if (a != 0.0f) atomicAdd(&avg_glob[i], a);
  }

  if (tid < 64) {
    int row = r0 + tid;
    if (rmask_s[tid] != 0) {
      idx_out[row] = (float)besti;
      float msep = 0.0f;
      for (int d = 0; d < 768; ++d) {
        float qh = emb[(size_t)besti * 768 + d];
        quant_out[(size_t)row * 768 + d] = qh;   // quantized = h + (q_hard - h) = q_hard
        float df = qh - hq[(size_t)row * 768 + d];
        msep += df * df;
      }
      atomicAdd(mse_glob, msep);
    } else {
      idx_out[row] = 0.0f;
      for (int d = 0; d < 768; ++d) quant_out[(size_t)row * 768 + d] = 0.0f;
    }
  }
}

__global__ void finalize_kernel(const float* __restrict__ avg_glob,
                                const float* __restrict__ mse_glob,
                                const float* __restrict__ nvalid,
                                float* __restrict__ out2)
{
  __shared__ float red[256];
  int tid = threadIdx.x;
  float nv = nvalid[0];
  float e = 0.0f;
  for (int i = tid; i < 8192; i += 256) {
    float a = avg_glob[i] / nv;
    e -= a * logf(a + 1e-10f);
  }
  red[tid] = e; __syncthreads();
  for (int off = 128; off > 0; off >>= 1) { if (tid < off) red[tid] += red[tid + off]; __syncthreads(); }
  if (tid == 0) {
    float ent = red[0];
    float mse = mse_glob[0] / nv;
    out2[0] = 1.25f * mse - 0.01f * ent; // loss = mse + 0.25*mse - 0.01*entropy
    out2[1] = __expf(ent);               // perplexity
  }
}

// ---------------------------------------------------------------------------
// Host orchestration
// ---------------------------------------------------------------------------
extern "C" void kernel_launch(void* const* d_in, const int* in_sizes, int n_in,
                              void* d_out, int out_size, void* d_ws, size_t ws_size,
                              hipStream_t stream)
{
  (void)in_sizes; (void)n_in; (void)out_size; (void)ws_size;
  const float* x       = (const float*)d_in[0];
  const int*   amask   = (const int*)d_in[1];
  const float* proj_W  = (const float*)d_in[2];
  const float* proj_b  = (const float*)d_in[3];
  const float* Wq      = (const float*)d_in[4];
  const float* bq      = (const float*)d_in[5];
  const float* Wk      = (const float*)d_in[6];
  const float* bk      = (const float*)d_in[7];
  const float* Wv      = (const float*)d_in[8];
  const float* bvv     = (const float*)d_in[9];
  const float* Wo      = (const float*)d_in[10];
  const float* bo      = (const float*)d_in[11];
  const float* ln1_g   = (const float*)d_in[12];
  const float* ln1_b   = (const float*)d_in[13];
  const float* ln2_g   = (const float*)d_in[14];
  const float* ln2_b   = (const float*)d_in[15];
  const float* ff_W1   = (const float*)d_in[16];
  const float* ff_b1   = (const float*)d_in[17];
  const float* ff_W2   = (const float*)d_in[18];
  const float* ff_b2   = (const float*)d_in[19];
  const float* pre_g   = (const float*)d_in[20];
  const float* pre_b   = (const float*)d_in[21];
  const float* emb     = (const float*)d_in[22];
  const float* qW      = (const float*)d_in[23];
  const float* qb      = (const float*)d_in[24];
  const float* kW      = (const float*)d_in[25];
  const float* kb      = (const float*)d_in[26];
  const float* temp    = (const float*)d_in[27];

  const int M = 8192;          // B*S
  char* ws = (char*)d_ws;
  size_t off = 0;
  auto alloc = [&](size_t bytes) -> void* {
    void* p = ws + off;
    off += (bytes + 255) & ~(size_t)255;
    return p;
  };

  // weights stored TRANSPOSED ([N,K] f16) for vectorized GEMM B staging
  _Float16* x16    = (_Float16*)alloc((size_t)M * 512 * 2);
  _Float16* wprojT = (_Float16*)alloc((size_t)512 * 768 * 2);
  _Float16* wqT    = (_Float16*)alloc((size_t)3 * 768 * 768 * 2);
  _Float16* wkT    = (_Float16*)alloc((size_t)3 * 768 * 768 * 2);
  _Float16* wvT    = (_Float16*)alloc((size_t)3 * 768 * 768 * 2);
  _Float16* woT    = (_Float16*)alloc((size_t)3 * 768 * 768 * 2);
  _Float16* w1T    = (_Float16*)alloc((size_t)3 * 768 * 3072 * 2);
  _Float16* w2T    = (_Float16*)alloc((size_t)3 * 3072 * 768 * 2);
  _Float16* wqWT   = (_Float16*)alloc((size_t)768 * 768 * 2);
  _Float16* wkWT   = (_Float16*)alloc((size_t)768 * 768 * 2);
  _Float16* emb16  = (_Float16*)alloc((size_t)8192 * 768 * 2);
  float*    h32    = (float*)   alloc((size_t)M * 768 * 4);
  _Float16* h16    = (_Float16*)alloc((size_t)M * 768 * 2);
  _Float16* q16    = (_Float16*)alloc((size_t)M * 768 * 2);
  _Float16* k16    = (_Float16*)alloc((size_t)M * 768 * 2);
  _Float16* vT16   = (_Float16*)alloc((size_t)M * 768 * 2);   // [768][8192] transposed V
  _Float16* ctx16  = (_Float16*)alloc((size_t)M * 768 * 2);
  float*    tmp32  = (float*)   alloc((size_t)M * 768 * 4);
  _Float16* ff1_16 = (_Float16*)alloc((size_t)M * 3072 * 2);
  _Float16* qry16  = (_Float16*)alloc((size_t)M * 768 * 2);
  _Float16* key16  = (_Float16*)alloc((size_t)M * 768 * 2);
  float*    avgp   = (float*)   alloc((size_t)8192 * 4);
  float*    msea   = (float*)   alloc(4);
  float*    nval   = (float*)   alloc(4);

  dim3 blk(256);
  auto cgrid = [](size_t n) { return dim3((unsigned)((n + 255) / 256)); };
  auto tgrid = [](size_t kn, unsigned l) { return dim3((unsigned)((kn + 255) / 256), l); };

  // --- precision conversion: activations/codebook (row-major), weights (transposed) ---
  cvt_f32_f16<<<cgrid((size_t)M * 512), blk, 0, stream>>>(x, x16, M * 512);
  cvt_f32_f16<<<cgrid((size_t)8192 * 768), blk, 0, stream>>>(emb, emb16, 8192 * 768);
  transpose_f32_f16<<<tgrid(512 * 768, 1), blk, 0, stream>>>(proj_W, wprojT, 512, 768);
  transpose_f32_f16<<<tgrid(768 * 768, 3), blk, 0, stream>>>(Wq, wqT, 768, 768);
  transpose_f32_f16<<<tgrid(768 * 768, 3), blk, 0, stream>>>(Wk, wkT, 768, 768);
  transpose_f32_f16<<<tgrid(768 * 768, 3), blk, 0, stream>>>(Wv, wvT, 768, 768);
  transpose_f32_f16<<<tgrid(768 * 768, 3), blk, 0, stream>>>(Wo, woT, 768, 768);
  transpose_f32_f16<<<tgrid((size_t)768 * 3072, 3), blk, 0, stream>>>(ff_W1, w1T, 768, 3072);
  transpose_f32_f16<<<tgrid((size_t)3072 * 768, 3), blk, 0, stream>>>(ff_W2, w2T, 3072, 768);
  transpose_f32_f16<<<tgrid(768 * 768, 1), blk, 0, stream>>>(qW, wqWT, 768, 768);
  transpose_f32_f16<<<tgrid(768 * 768, 1), blk, 0, stream>>>(kW, wkWT, 768, 768);
  nvalid_kernel<<<1, blk, 0, stream>>>(amask, nval);

  // --- input projection: h = x @ proj_W + proj_b ---
  gemm_f16_kernel<<<dim3(3, 128), blk, 0, stream>>>(x16, wprojT, proj_b, h32, h16, nullptr, M, 768, 512, 0);

  // --- transformer layers ---
  for (int i = 0; i < 3; ++i) {
    const size_t woff = (size_t)i * 768 * 768;
    gemm_f16_kernel<<<dim3(3, 128), blk, 0, stream>>>(h16, wqT + woff, bq + i * 768, nullptr, q16, nullptr, M, 768, 768, 0);
    gemm_f16_kernel<<<dim3(3, 128), blk, 0, stream>>>(h16, wkT + woff, bk + i * 768, nullptr, k16, nullptr, M, 768, 768, 0);
    gemm_f16_kernel<<<dim3(3, 128), blk, 0, stream>>>(h16, wvT + woff, bvv + i * 768, nullptr, nullptr, vT16, M, 768, 768, 0);
    flash_attn_kernel<<<dim3(16, 8, 8), blk, 0, stream>>>(q16, k16, vT16, amask, ctx16);
    gemm_f16_kernel<<<dim3(3, 128), blk, 0, stream>>>(ctx16, woT + woff, bo + i * 768, tmp32, nullptr, nullptr, M, 768, 768, 0);
    ln_kernel<<<8192, blk, 0, stream>>>(h32, tmp32, ln1_g + i * 768, ln1_b + i * 768, h32, h16);
    gemm_f16_kernel<<<dim3(12, 128), blk, 0, stream>>>(h16, w1T + (size_t)i * 768 * 3072, ff_b1 + i * 3072, nullptr, ff1_16, nullptr, M, 3072, 768, 1);
    gemm_f16_kernel<<<dim3(3, 128), blk, 0, stream>>>(ff1_16, w2T + (size_t)i * 3072 * 768, ff_b2 + i * 768, tmp32, nullptr, nullptr, M, 768, 3072, 0);
    ln_kernel<<<8192, blk, 0, stream>>>(h32, tmp32, ln2_g + i * 768, ln2_b + i * 768, h32, h16);
  }

  // --- final LN + VQ projections ---
  ln_kernel<<<8192, blk, 0, stream>>>(h32, nullptr, pre_g, pre_b, h32, h16);
  gemm_f16_kernel<<<dim3(3, 128), blk, 0, stream>>>(h16, wqWT, qb, nullptr, qry16, nullptr, M, 768, 768, 0);
  gemm_f16_kernel<<<dim3(3, 128), blk, 0, stream>>>(emb16, wkWT, kb, nullptr, key16, nullptr, 8192, 768, 768, 0);

  // --- VQ: argmax / softmax stats / quantize / mse ---
  zero_f32<<<cgrid(8192), blk, 0, stream>>>(avgp, 8192);
  zero_f32<<<1, blk, 0, stream>>>(msea, 1);
  float* out = (float*)d_out;
  float* quant = out;                      // [8,1024,768]
  float* scal  = out + (size_t)M * 768;    // loss, perplexity
  float* idxo  = scal + 2;                 // [8,1024]
  vq_kernel<<<128, blk, 0, stream>>>(qry16, key16, h32, emb, amask, temp, quant, idxo, avgp, msea);
  finalize_kernel<<<1, blk, 0, stream>>>(avgp, msea, nval, scal);
}
